// HeterogeneousGATLayer_74174085202173
// MI455X (gfx1250) — matrix operand order repaired and checked by reading, not compiled, not run on hardware
//
#include <hip/hip_runtime.h>

// Problem constants (match reference)
#define NNODES 10000
#define MPAD   10112          // 79 * 128, zero-padded rows for guard-free GEMM
#define DFEAT  512
#define NHEAD  4
#define CHID   128
#define NEDGE  100000
#define HC     512            // NHEAD*CHID
#define ETOT   (NEDGE + NNODES)

#if __has_builtin(__builtin_amdgcn_global_load_async_to_lds_b128) && \
    __has_builtin(__builtin_amdgcn_s_wait_asynccnt)
#define USE_ASYNC_LDS 1
#else
#define USE_ASYNC_LDS 0
#endif

// ---------------- types for WMMA / async ----------------
typedef __attribute__((ext_vector_type(8)))  float  v8f;
typedef __attribute__((ext_vector_type(8)))  __bf16 v8bf;
typedef __attribute__((ext_vector_type(16))) __bf16 v16bf;

// builtin parameter type per hipcc diagnostic: vector_size(16) int, AS(1)/AS(3)
typedef int v4i_vs __attribute__((vector_size(16)));
typedef __attribute__((address_space(1))) v4i_vs* gbl_v4i_p;
typedef __attribute__((address_space(3))) v4i_vs* lds_v4i_p;

union BF16x16 { v16bf v; v8bf h[2]; };

__device__ __forceinline__ unsigned short f32_to_bf16(float f) {
  unsigned u = __float_as_uint(f);
  unsigned lsb = (u >> 16) & 1u;
  u += 0x7fffu + lsb;                 // round-to-nearest-even
  return (unsigned short)(u >> 16);
}

// order-preserving map float -> uint (for atomicMax-based segment max)
__device__ __forceinline__ unsigned f2ord(float f) {
  unsigned u = __float_as_uint(f);
  return (u & 0x80000000u) ? ~u : (u | 0x80000000u);
}
__device__ __forceinline__ float ord2f(unsigned s) {
  unsigned u = (s & 0x80000000u) ? (s & 0x7fffffffu) : ~s;
  return __uint_as_float(u);
}

// ---------------- K0: x[N][512] f32 -> xb[MPAD][512] bf16 (zero padded) ----------
__global__ __launch_bounds__(256) void cvt_x_bf16(
    const float* __restrict__ x, unsigned short* __restrict__ xb) {
  int i = blockIdx.x * blockDim.x + threadIdx.x;   // quad index
  size_t base = (size_t)i * 4;
  int row = (int)(base >> 9);
  float4 v = {0.f, 0.f, 0.f, 0.f};
  if (row < NNODES) v = *(const float4*)(x + base);
  ushort4 q;
  q.x = f32_to_bf16(v.x); q.y = f32_to_bf16(v.y);
  q.z = f32_to_bf16(v.z); q.w = f32_to_bf16(v.w);
  *(ushort4*)(xb + base) = q;
}

// ---------------- K1: W[D][HC] f32 -> Wt[HC][D] bf16 (transposed) ----------------
__global__ __launch_bounds__(256) void transpose_cvt_bf16(
    const float* __restrict__ W, unsigned short* __restrict__ Wt) {
  __shared__ float tile[16][17];
  int k0 = blockIdx.x * 16, n0 = blockIdx.y * 16;
  int tx = threadIdx.x & 15, ty = threadIdx.x >> 4;
  tile[ty][tx] = W[(size_t)(k0 + ty) * HC + n0 + tx];
  __syncthreads();
  Wt[(size_t)(n0 + ty) * DFEAT + k0 + tx] = f32_to_bf16(tile[tx][ty]);
}

// ---------------- K2: C[M][512] = A(bf16,[MPAD][512]) x Wt(bf16,[n][k]) ----------
// Block tile 128x128x32, 256 threads = 8 waves (4 M x 2 N), wave tile 32x64.
// Both tiles staged to LDS with async global->LDS DMA when available.
__global__ __launch_bounds__(256) void gemm_wmma_bf16(
    const unsigned short* __restrict__ A, const unsigned short* __restrict__ Bt,
    float* __restrict__ C, int M) {
  __shared__ unsigned short As[128 * 40];   // +8 bf16 skew per row (80B stride)
  __shared__ unsigned short Bs[128 * 40];

  const int tid  = threadIdx.x;
  const int lane = tid & 31;
  const int wid  = tid >> 5;
  const int wm   = wid & 3;        // 0..3 -> M strip of 32
  const int wn   = wid >> 2;       // 0..1 -> N strip of 64
  const int l16  = lane & 15;
  const int lh   = lane >> 4;      // lane half

  const int m0 = blockIdx.x * 128;
  const int n0 = blockIdx.y * 128;

  v8f acc[2][4];
  const v8f vz = {0.f, 0.f, 0.f, 0.f, 0.f, 0.f, 0.f, 0.f};
#pragma unroll
  for (int i = 0; i < 2; ++i)
#pragma unroll
    for (int j = 0; j < 4; ++j) acc[i][j] = vz;

  const int r2 = tid >> 2;            // 0..63  (row within half-tile)
  const int kc = (tid & 3) * 8;       // bf16 column: 0,8,16,24 -> 16B chunks

  for (int kt = 0; kt < DFEAT / 32; ++kt) {
    const int k0 = kt * 32;
    // stage A + B tiles: each thread moves 2x16B per matrix, no bounds checks
#pragma unroll
    for (int i = 0; i < 2; ++i) {
      int row = r2 + i * 64;
      const unsigned short* gA = A + (size_t)(m0 + row) * DFEAT + k0 + kc;
      const unsigned short* gB = Bt + (size_t)(n0 + row) * DFEAT + k0 + kc;
      unsigned short* lA = &As[row * 40 + kc];
      unsigned short* lB = &Bs[row * 40 + kc];
#if USE_ASYNC_LDS
      __builtin_amdgcn_global_load_async_to_lds_b128(
          (gbl_v4i_p)gA, (lds_v4i_p)lA, 0, 0);
      __builtin_amdgcn_global_load_async_to_lds_b128(
          (gbl_v4i_p)gB, (lds_v4i_p)lB, 0, 0);
#else
      *(uint4*)lA = *(const uint4*)gA;
      *(uint4*)lB = *(const uint4*)gB;
#endif
    }
#if USE_ASYNC_LDS
    __builtin_amdgcn_s_wait_asynccnt(0);
#endif
    __syncthreads();

    // fragments per ISA 16-bit layouts
    BF16x16 afr[2], bfr[4];
#pragma unroll
    for (int mt = 0; mt < 2; ++mt) {
      int row = wm * 32 + mt * 16 + l16;
      const unsigned short* p = &As[row * 40 + lh * 8];
      afr[mt].h[0] = *(const v8bf*)(p);        // K = lh*8 .. +7
      afr[mt].h[1] = *(const v8bf*)(p + 16);   // K = lh*8+16 .. +7
    }
#pragma unroll
    for (int nt = 0; nt < 4; ++nt) {
      int col = wn * 64 + nt * 16 + l16;
      const unsigned short* p = &Bs[col * 40 + lh * 8];
      bfr[nt].h[0] = *(const v8bf*)(p);        // K = lh*8 .. +7
      bfr[nt].h[1] = *(const v8bf*)(p + 16);   // K = lh*8+16 .. +7
    }
#pragma unroll
    for (int mt = 0; mt < 2; ++mt)
#pragma unroll
      for (int nt = 0; nt < 4; ++nt)
        acc[mt][nt] = __builtin_amdgcn_wmma_f32_16x16x32_bf16(
            false, afr[mt].v, false, bfr[nt].v, (short)0, acc[mt][nt],
            false, false);
    __syncthreads();
  }

  // store: acc element r -> (M = r + 8*lh, N = l16) within 16x16 tile
#pragma unroll
  for (int mt = 0; mt < 2; ++mt) {
    int mbase = m0 + wm * 32 + mt * 16 + lh * 8;
#pragma unroll
    for (int nt = 0; nt < 4; ++nt) {
      int col = n0 + wn * 64 + nt * 16 + l16;
#pragma unroll
      for (int r = 0; r < 8; ++r) {
        int m = mbase + r;
        if (m < M) C[(size_t)m * HC + col] = acc[mt][nt][r];
      }
    }
  }
}

// ---------------- K3: a[n][h] = dot(h[n, h*128 : +128], att[h]) ------------------
__global__ __launch_bounds__(256) void att_scores(
    const float* __restrict__ hfeat, const float* __restrict__ att,
    float* __restrict__ out) {
  int gw   = (blockIdx.x * blockDim.x + threadIdx.x) >> 5;
  int lane = threadIdx.x & 31;
  int n  = gw >> 2;
  int hd = gw & 3;
  const float* hp = hfeat + (size_t)n * HC + hd * CHID;
  const float* ap = att + hd * CHID;
  float s = 0.f;
#pragma unroll
  for (int c = lane; c < CHID; c += 32) s += hp[c] * ap[c];
#pragma unroll
  for (int off = 16; off > 0; off >>= 1) s += __shfl_xor(s, off, 32);
  if (lane == 0) out[n * NHEAD + hd] = s;
}

// ---------------- K4: zero accum/denom, reset ordered-max ------------------------
__global__ __launch_bounds__(256) void init_bufs(
    unsigned* __restrict__ amaxbits, float* __restrict__ denom,
    float* __restrict__ accum) {
  int i = blockIdx.x * blockDim.x + threadIdx.x;
  if (i < NNODES * NHEAD) { amaxbits[i] = 0u; denom[i] = 0.f; }
  if (i < NNODES * HC) accum[i] = 0.f;
}

// ---------------- K5: alpha = leaky_relu(a_src[s]+a_dst[d]); segment max ---------
__global__ __launch_bounds__(256) void edge_alpha(
    const int* __restrict__ ei, const float* __restrict__ asrc,
    const float* __restrict__ adst, float* __restrict__ exbuf,
    unsigned* __restrict__ amaxbits) {
  int e = blockIdx.x * blockDim.x + threadIdx.x;
  if (e >= ETOT) return;
  int s, d;
  if (e < NEDGE) { s = ei[e]; d = ei[NEDGE + e]; } else { s = d = e - NEDGE; }
#pragma unroll
  for (int hd = 0; hd < NHEAD; ++hd) {
    float a = asrc[s * NHEAD + hd] + adst[d * NHEAD + hd];
    a = (a > 0.f) ? a : 0.2f * a;           // negative_slope = 0.2
    exbuf[(size_t)e * NHEAD + hd] = a;
    atomicMax(&amaxbits[d * NHEAD + hd], f2ord(a));
  }
}

// ---------------- K6: ex = exp(alpha - amax[d]); segment sum ---------------------
__global__ __launch_bounds__(256) void edge_norm(
    const int* __restrict__ ei, const unsigned* __restrict__ amaxbits,
    float* __restrict__ exbuf, float* __restrict__ denom) {
  int e = blockIdx.x * blockDim.x + threadIdx.x;
  if (e >= ETOT) return;
  int d = (e < NEDGE) ? ei[NEDGE + e] : (e - NEDGE);
#pragma unroll
  for (int hd = 0; hd < NHEAD; ++hd) {
    float a  = exbuf[(size_t)e * NHEAD + hd];
    float mx = ord2f(amaxbits[d * NHEAD + hd]);
    float ex = __expf(a - mx);
    exbuf[(size_t)e * NHEAD + hd] = ex;
    atomicAdd(&denom[d * NHEAD + hd], ex);
  }
}

// ---------------- K7: accum[d] += h_src[s] * coef (block per edge) ---------------
__global__ __launch_bounds__(256) void edge_scatter(
    const int* __restrict__ ei, const float* __restrict__ exbuf,
    const float* __restrict__ denom, const float* __restrict__ hsrc,
    float* __restrict__ accum) {
  __shared__ float scoef[NHEAD];
  int e = blockIdx.x;
  int s, d;
  if (e < NEDGE) { s = ei[e]; d = ei[NEDGE + e]; } else { s = d = e - NEDGE; }
  if (threadIdx.x < NHEAD) {
    scoef[threadIdx.x] = exbuf[(size_t)e * NHEAD + threadIdx.x] /
                         (denom[d * NHEAD + threadIdx.x] + 1e-16f);
  }
  __syncthreads();
  const float* hp = hsrc + (size_t)s * HC;
  float* op = accum + (size_t)d * HC;
#pragma unroll
  for (int j = threadIdx.x; j < HC; j += 256) {
    atomicAdd(&op[j], hp[j] * scoef[j >> 7]);
  }
}

// ---------------- K8: head mean + bias ------------------------------------------
__global__ __launch_bounds__(256) void finalize(
    const float* __restrict__ accum, const float* __restrict__ bias,
    float* __restrict__ out) {
  int i = blockIdx.x * blockDim.x + threadIdx.x;
  if (i >= NNODES * CHID) return;
  int n = i >> 7, c = i & 127;
  const float* p = accum + (size_t)n * HC;
  out[i] = 0.25f * (p[c] + p[CHID + c] + p[2 * CHID + c] + p[3 * CHID + c]) +
           bias[c];
}

// ---------------- host launcher --------------------------------------------------
extern "C" void kernel_launch(void* const* d_in, const int* in_sizes, int n_in,
                              void* d_out, int out_size, void* d_ws,
                              size_t ws_size, hipStream_t stream) {
  (void)in_sizes; (void)n_in; (void)out_size; (void)ws_size;

  const float* xs[3] = {(const float*)d_in[0], (const float*)d_in[1],
                        (const float*)d_in[2]};
  // conv order matches reference compute order: a->b, b->c, c->a
  const int srcs[3] = {0, 1, 2};
  const int dsts[3] = {1, 2, 0};
  const int wb[3]   = {3, 8, 13};    // Ws, Wd, as, ad, b per conv
  const int eii[3]  = {18, 19, 20};
  const int osl[3]  = {1, 2, 0};     // out slot: out_b, out_c, out_a

  // workspace carve-out (256B aligned)
  char* ws = (char*)d_ws;
  size_t off = 0;
  auto alc = [&](size_t bytes) {
    size_t o = off;
    off += (bytes + 255) & ~(size_t)255;
    return o;
  };
  unsigned short* xb[3];
  for (int i = 0; i < 3; ++i)
    xb[i] = (unsigned short*)(ws + alc((size_t)MPAD * DFEAT * 2));
  float*          hsrc     = (float*)(ws + alc((size_t)NNODES * HC * 4));
  float*          hdst     = (float*)(ws + alc((size_t)NNODES * HC * 4));
  unsigned short* Wt       = (unsigned short*)(ws + alc((size_t)HC * DFEAT * 2));
  float*          a_src    = (float*)(ws + alc((size_t)NNODES * NHEAD * 4));
  float*          a_dst    = (float*)(ws + alc((size_t)NNODES * NHEAD * 4));
  unsigned*       amaxbits = (unsigned*)(ws + alc((size_t)NNODES * NHEAD * 4));
  float*          denom    = (float*)(ws + alc((size_t)NNODES * NHEAD * 4));
  float*          exbuf    = (float*)(ws + alc((size_t)ETOT * NHEAD * 4));
  float*          accum    = (float*)(ws + alc((size_t)NNODES * HC * 4));

  float* outp = (float*)d_out;

  const dim3 tblk(256);
  const dim3 tgrid(DFEAT / 16, HC / 16);             // transpose grid
  const dim3 ggrid(MPAD / 128, HC / 128);            // 79 x 4
  const int  cvt_blocks  = (MPAD * DFEAT / 4) / 256; // exact
  const int  att_blocks  = (NNODES * NHEAD) / 8;     // 8 waves/block
  const int  init_blocks = (NNODES * HC + 255) / 256;
  const int  edge_blocks = (ETOT + 255) / 256;
  const int  fin_blocks  = (NNODES * CHID + 255) / 256;

  // one-time bf16 conversion of node features (each reused by two convs)
  for (int i = 0; i < 3; ++i)
    cvt_x_bf16<<<cvt_blocks, tblk, 0, stream>>>(xs[i], xb[i]);

  for (int cv = 0; cv < 3; ++cv) {
    const float* Wsrc = (const float*)d_in[wb[cv] + 0];
    const float* Wdst = (const float*)d_in[wb[cv] + 1];
    const float* atts = (const float*)d_in[wb[cv] + 2];
    const float* attd = (const float*)d_in[wb[cv] + 3];
    const float* bias = (const float*)d_in[wb[cv] + 4];
    const int*   ei   = (const int*)d_in[eii[cv]];
    float* outsec = outp + (size_t)osl[cv] * NNODES * CHID;

    transpose_cvt_bf16<<<tgrid, tblk, 0, stream>>>(Wsrc, Wt);
    gemm_wmma_bf16<<<ggrid, tblk, 0, stream>>>(xb[srcs[cv]], Wt, hsrc, NNODES);
    transpose_cvt_bf16<<<tgrid, tblk, 0, stream>>>(Wdst, Wt);
    gemm_wmma_bf16<<<ggrid, tblk, 0, stream>>>(xb[dsts[cv]], Wt, hdst, NNODES);

    att_scores<<<att_blocks, tblk, 0, stream>>>(hsrc, atts, a_src);
    att_scores<<<att_blocks, tblk, 0, stream>>>(hdst, attd, a_dst);

    init_bufs<<<init_blocks, tblk, 0, stream>>>(amaxbits, denom, accum);
    edge_alpha<<<edge_blocks, tblk, 0, stream>>>(ei, a_src, a_dst, exbuf,
                                                 amaxbits);
    edge_norm<<<edge_blocks, tblk, 0, stream>>>(ei, amaxbits, exbuf, denom);
    edge_scatter<<<ETOT, tblk, 0, stream>>>(ei, exbuf, denom, hsrc, accum);
    finalize<<<fin_blocks, tblk, 0, stream>>>(accum, bias, outsec);
  }
}
